// AttentionModule_39462159515861
// MI455X (gfx1250) — matrix operand order repaired and checked by reading, not compiled, not run on hardware
//
#include <hip/hip_runtime.h>

typedef __bf16 bf16;
typedef __attribute__((ext_vector_type(16))) __bf16 v16bf;
typedef __attribute__((ext_vector_type(8)))  __bf16 v8bf;
typedef __attribute__((ext_vector_type(8)))  float  v8f;
typedef __attribute__((ext_vector_type(4)))  unsigned int uint32x4;
typedef __attribute__((ext_vector_type(8)))  int int32x8;
typedef __attribute__((ext_vector_type(4)))  int int32x4;

#define BATCH   4
#define NSEQ    2048
#define DIM_    1024
#define NHEAD   16
#define HD      64
#define ROWS    (BATCH * NSEQ)   /* 8192 */
#define NQKV    (3 * DIM_)       /* 3072 */

// ---------------------------------------------------------------------------
// helpers: WMMA fragment loads, contiguous k order (k = 16*h + s) in both A and B
// ---------------------------------------------------------------------------
__device__ __forceinline__ v16bf concat8(v8bf lo, v8bf hi) {
  return __builtin_shufflevector(lo, hi, 0,1,2,3,4,5,6,7,8,9,10,11,12,13,14,15);
}

// A-matrix 16x32 bf16 fragment for one row `rowPtr` (lane = m + 16*h):
//   elements 0..7  : k = k0 + 16*h + e
//   elements 8..15 : k = k0 + 16*h + 8 + e
__device__ __forceinline__ v16bf load_a_frag(const bf16* rowPtr, int k0, int h) {
  v8bf lo = *(const v8bf*)(rowPtr + k0 + 16 * h);
  v8bf hi = *(const v8bf*)(rowPtr + k0 + 16 * h + 8);
  return concat8(lo, hi);
}

__device__ __forceinline__ v8f wmma_bf16(v16bf a, v16bf b, v8f c) {
  return __builtin_amdgcn_wmma_f32_16x16x32_bf16(false, a, false, b, (short)0, c,
                                                 false, false);
}

// ---------------------------------------------------------------------------
// fp32 -> bf16 conversion (8 elements / thread, b128 load + b128 store)
// ---------------------------------------------------------------------------
__global__ __launch_bounds__(256) void cvt_f32_bf16(const float* __restrict__ in,
                                                    bf16* __restrict__ out, int n8) {
  int i = blockIdx.x * blockDim.x + threadIdx.x;
  if (i >= n8) return;
  const float4* p = (const float4*)in + (size_t)i * 2;
  float4 a = p[0], b = p[1];
  v8bf o;
  o[0] = (bf16)a.x; o[1] = (bf16)a.y; o[2] = (bf16)a.z; o[3] = (bf16)a.w;
  o[4] = (bf16)b.x; o[5] = (bf16)b.y; o[6] = (bf16)b.z; o[7] = (bf16)b.w;
  ((v8bf*)out)[i] = o;
}

// ---------------------------------------------------------------------------
// QKV GEMM: [8192 x 1024] @ W^T[1024 x 3072] + b, scatter into Q/K/V [B,H,N,64]
// 8 waves/block, wave tile 32x64 (2x4 WMMA tiles), block tile 64x256.
// ---------------------------------------------------------------------------
__global__ __launch_bounds__(256) void qkv_gemm(const bf16* __restrict__ X,
                                                const bf16* __restrict__ W,
                                                const float* __restrict__ bias,
                                                bf16* __restrict__ Q,
                                                bf16* __restrict__ K,
                                                bf16* __restrict__ V) {
  const int lane  = threadIdx.x & 31;
  const int wave  = threadIdx.x >> 5;
  const int waveM = wave >> 2, waveN = wave & 3;
  const int h = lane >> 4, l16 = lane & 15;
  const int blockM = blockIdx.x * 64;
  const int blockN = blockIdx.y * 256;

  v8f acc[2][4] = {};

  const bf16* arow0 = X + (size_t)(blockM + waveM * 32 + l16) * DIM_;
  const bf16* arow1 = arow0 + 16 * DIM_;
  const bf16* brow[4];
#pragma unroll
  for (int nt = 0; nt < 4; ++nt)
    brow[nt] = W + (size_t)(blockN + waveN * 64 + nt * 16 + l16) * DIM_;

  for (int k0 = 0; k0 < DIM_; k0 += 32) {
    if (k0 + 128 < DIM_) {   // stream-prefetch A operand -> global_prefetch_b8
      __builtin_prefetch(arow0 + k0 + 128, 0, 3);
      __builtin_prefetch(arow1 + k0 + 128, 0, 3);
    }
    v16bf a0 = load_a_frag(arow0, k0, h);
    v16bf a1 = load_a_frag(arow1, k0, h);
#pragma unroll
    for (int nt = 0; nt < 4; ++nt) {
      // B-matrix 32x16: lane = n + 16*(k/16), element = k%16 -> 32B contiguous
      v16bf b = *(const v16bf*)(brow[nt] + k0 + 16 * h);
      acc[0][nt] = wmma_bf16(a0, b, acc[0][nt]);
      acc[1][nt] = wmma_bf16(a1, b, acc[1][nt]);
    }
  }

  const float qscale = 0.125f;  // HEAD_DIM^-0.5, folded into Q
#pragma unroll
  for (int mt = 0; mt < 2; ++mt) {
#pragma unroll
    for (int nt = 0; nt < 4; ++nt) {
      int col   = blockN + waveN * 64 + nt * 16 + l16;
      int which = col >> 10;            // 0=Q 1=K 2=V (uniform per wave tile)
      int hh    = (col >> 6) & 15;
      int d     = col & 63;
      bf16* dst = (which == 0) ? Q : (which == 1) ? K : V;
      float bv  = bias[col];
#pragma unroll
      for (int i = 0; i < 8; ++i) {
        int row = blockM + waveM * 32 + mt * 16 + 8 * h + i;  // C-layout row
        int b   = row >> 11, n = row & 2047;
        float v = acc[mt][nt][i] + bv;
        if (which == 0) v *= qscale;
        dst[(((size_t)(b * NHEAD + hh) * NSEQ) + n) * HD + d] = (bf16)v;
      }
    }
  }
}

// ---------------------------------------------------------------------------
// Flash attention: block = 4 waves (128 thr) handles (b,h, 64 q-rows).
// Wave owns 16 q-rows; loop over 64-key blocks with online softmax.
// K tile is staged into LDS by the Tensor Data Mover (tensor_load_to_lds),
// overlapped with the VALU/LSU V-transpose staging; tracked via TENSORcnt.
// ---------------------------------------------------------------------------
__global__ __launch_bounds__(128) void attn_kernel(const bf16* __restrict__ Q,
                                                   const bf16* __restrict__ K,
                                                   const bf16* __restrict__ V,
                                                   bf16* __restrict__ O) {
  __shared__ bf16 Kt[64][64];        // K block, row-major (TDM destination)
  __shared__ bf16 VT[64][80];        // V transposed: VT[d][key], padded rows
  __shared__ bf16 Pls[4][16][72];    // per-wave P tile: [m][key]

  const int lane = threadIdx.x & 31;
  const int wave = threadIdx.x >> 5;
  const int h = lane >> 4, l16 = lane & 15;
  const int bh = blockIdx.y;                     // b*16 + head
  const size_t base = (size_t)bh * NSEQ * HD;
  const bf16* Qb = Q + base;
  const bf16* Kb = K + base;
  const bf16* Vb = V + base;
  const int qrow = blockIdx.x * 64 + wave * 16;

  // Q fragments (A layout), d-chunks [0,32) and [32,64); Q pre-scaled.
  const bf16* qr = Qb + (size_t)(qrow + l16) * HD;
  v16bf qa0 = load_a_frag(qr, 0, h);
  v16bf qa1 = load_a_frag(qr, 32, h);

  v8f o[4] = {};
  float m_i[8], l_i[8];
#pragma unroll
  for (int i = 0; i < 8; ++i) { m_i[i] = -__builtin_inff(); l_i[i] = 0.0f; }

  const unsigned kt_lds = (unsigned)(unsigned long long)(void*)&Kt[0][0];

  for (int kb = 0; kb < NSEQ; kb += 64) {
    __syncthreads();  // previous iteration's LDS readers are done

    // ---- TDM: DMA the 64x64 bf16 K block into LDS (D# per ISA 8.3/8.4) ----
    if (wave == 0) {
      unsigned long long ga = (unsigned long long)(const void*)(Kb + (size_t)kb * HD);
      uint32x4 g0;
      g0[0] = 1u;                                        // count=1 (valid D#)
      g0[1] = kt_lds;                                    // lds_addr (bytes)
      g0[2] = (unsigned)ga;                              // global_addr[31:0]
      g0[3] = (unsigned)((ga >> 32) & 0x1FFFFFFu) | (2u << 30);  // [56:32] | type=2
      int32x8 g1 = {};
      g1[0] = 1 << 16;              // data_size=1 (2 bytes); wg_mask=0; no pad
      g1[1] = 64 << 16;             // tensor_dim0[15:0]  = 64  (bits 79:48)
      g1[2] = (int)(2048u << 16);   // tensor_dim1[15:0]  = 2048 (bits 111:80)
      g1[3] = 64 << 16;             // tile_dim0 = 64            (bits 127:112)
      g1[4] = 64;                   // tile_dim1 = 64            (bits 143:128)
      g1[5] = 64;                   // tensor_dim0_stride = 64   (bits 207:160)
      int32x4 gz4 = {};
      int32x8 gz8 = {};
      // amdgpu-toolchain (clang-23) 6-arg form: (g0, g1, g2, g3, g4, cpol)
      __builtin_amdgcn_tensor_load_to_lds(g0, g1, gz4, gz4, gz8, 0);
    }

    // ---- stage V block transposed into LDS (cooperative, 128 threads) ----
    {
      int t = threadIdx.x;
      int key = t >> 1;
      int dBase = (t & 1) * 32;
      const v8bf* src = (const v8bf*)(Vb + (size_t)(kb + key) * HD + dBase);
#pragma unroll
      for (int c = 0; c < 4; ++c) {
        v8bf vv = src[c];
#pragma unroll
        for (int e = 0; e < 8; ++e) VT[dBase + c * 8 + e][key] = vv[e];
      }
    }
    if (wave == 0) __builtin_amdgcn_s_wait_tensorcnt(0);
    __syncthreads();

    // ---- S = Qs . K^T  (16 x 64), B frags from the TDM-staged K tile ----
    v8f s[4];
#pragma unroll
    for (int nt = 0; nt < 4; ++nt) {
      v8f sa = {};
      const bf16* krow = &Kt[nt * 16 + l16][0];
      v16bf bk0 = *(const v16bf*)(krow + 0 + 16 * h);
      v16bf bk1 = *(const v16bf*)(krow + 32 + 16 * h);
      sa = wmma_bf16(qa0, bk0, sa);
      sa = wmma_bf16(qa1, bk1, sa);
      s[nt] = sa;
    }

    // ---- online softmax (row r = 8*h + i lives on 16 lanes of a half) ----
    float alpha[8];
#pragma unroll
    for (int i = 0; i < 8; ++i) {
      float mx = fmaxf(fmaxf(s[0][i], s[1][i]), fmaxf(s[2][i], s[3][i]));
#pragma unroll
      for (int off = 8; off >= 1; off >>= 1) mx = fmaxf(mx, __shfl_xor(mx, off, 32));
      float mnew = fmaxf(m_i[i], mx);
      alpha[i] = __expf(m_i[i] - mnew);
      m_i[i] = mnew;
      float rsum = 0.0f;
#pragma unroll
      for (int nt = 0; nt < 4; ++nt) {
        float p = __expf(s[nt][i] - mnew);
        s[nt][i] = p;
        rsum += p;
      }
#pragma unroll
      for (int off = 8; off >= 1; off >>= 1) rsum += __shfl_xor(rsum, off, 32);
      l_i[i] = l_i[i] * alpha[i] + rsum;
    }
#pragma unroll
    for (int dt = 0; dt < 4; ++dt)
#pragma unroll
      for (int i = 0; i < 8; ++i) o[dt][i] = o[dt][i] * alpha[i];

    // ---- P: C layout -> LDS -> A layout (wave-private; LDS is in-order) ----
#pragma unroll
    for (int nt = 0; nt < 4; ++nt)
#pragma unroll
      for (int i = 0; i < 8; ++i)
        Pls[wave][8 * h + i][nt * 16 + l16] = (bf16)s[nt][i];
    asm volatile("s_wait_dscnt 0" ::: "memory");

    // ---- O += P . V  (contraction over 64 keys, 2 chunks of 32) ----
#pragma unroll
    for (int kt = 0; kt < 2; ++kt) {
      const bf16* prow = &Pls[wave][l16][0];
      v8bf plo = *(const v8bf*)(prow + kt * 32 + 16 * h);
      v8bf phi = *(const v8bf*)(prow + kt * 32 + 16 * h + 8);
      v16bf pa = concat8(plo, phi);
#pragma unroll
      for (int dt = 0; dt < 4; ++dt) {
        const bf16* vrow = &VT[dt * 16 + l16][0];
        v16bf bv = *(const v16bf*)(vrow + kt * 32 + 16 * h);  // 32B aligned
        o[dt] = wmma_bf16(pa, bv, o[dt]);
      }
    }
  }

  // ---- normalize and store into [B, N, H*64] bf16 for proj GEMM ----
  const int b = bh >> 4, hh = bh & 15;
#pragma unroll
  for (int dt = 0; dt < 4; ++dt) {
#pragma unroll
    for (int i = 0; i < 8; ++i) {
      int n = qrow + 8 * h + i;
      int d = dt * 16 + l16;
      float v = o[dt][i] / l_i[i];
      O[((size_t)(b * NSEQ + n) * DIM_) + hh * HD + d] = (bf16)v;
    }
  }
}

// ---------------------------------------------------------------------------
// Output projection: [8192 x 1024] @ Wp^T[1024 x 1024] + b -> fp32 out
// ---------------------------------------------------------------------------
__global__ __launch_bounds__(256) void proj_gemm(const bf16* __restrict__ A,
                                                 const bf16* __restrict__ W,
                                                 const float* __restrict__ bias,
                                                 float* __restrict__ out) {
  const int lane  = threadIdx.x & 31;
  const int wave  = threadIdx.x >> 5;
  const int waveM = wave >> 2, waveN = wave & 3;
  const int h = lane >> 4, l16 = lane & 15;
  const int blockM = blockIdx.x * 64;
  const int blockN = blockIdx.y * 256;

  v8f acc[2][4] = {};

  const bf16* arow0 = A + (size_t)(blockM + waveM * 32 + l16) * DIM_;
  const bf16* arow1 = arow0 + 16 * DIM_;
  const bf16* brow[4];
#pragma unroll
  for (int nt = 0; nt < 4; ++nt)
    brow[nt] = W + (size_t)(blockN + waveN * 64 + nt * 16 + l16) * DIM_;

  for (int k0 = 0; k0 < DIM_; k0 += 32) {
    if (k0 + 128 < DIM_) {
      __builtin_prefetch(arow0 + k0 + 128, 0, 3);
      __builtin_prefetch(arow1 + k0 + 128, 0, 3);
    }
    v16bf a0 = load_a_frag(arow0, k0, h);
    v16bf a1 = load_a_frag(arow1, k0, h);
#pragma unroll
    for (int nt = 0; nt < 4; ++nt) {
      v16bf b = *(const v16bf*)(brow[nt] + k0 + 16 * h);
      acc[0][nt] = wmma_bf16(a0, b, acc[0][nt]);
      acc[1][nt] = wmma_bf16(a1, b, acc[1][nt]);
    }
  }

#pragma unroll
  for (int mt = 0; mt < 2; ++mt) {
#pragma unroll
    for (int nt = 0; nt < 4; ++nt) {
      int col  = blockN + waveN * 64 + nt * 16 + l16;
      float bv = bias[col];
#pragma unroll
      for (int i = 0; i < 8; ++i) {
        int row = blockM + waveM * 32 + mt * 16 + 8 * h + i;
        out[(size_t)row * DIM_ + col] = acc[mt][nt][i] + bv;
      }
    }
  }
}

// ---------------------------------------------------------------------------
// launcher
// ---------------------------------------------------------------------------
extern "C" void kernel_launch(void* const* d_in, const int* in_sizes, int n_in,
                              void* d_out, int out_size, void* d_ws, size_t ws_size,
                              hipStream_t stream) {
  const float* x      = (const float*)d_in[0];
  const float* w_qkv  = (const float*)d_in[1];
  const float* b_qkv  = (const float*)d_in[2];
  const float* w_proj = (const float*)d_in[3];
  const float* b_proj = (const float*)d_in[4];
  float* out = (float*)d_out;

  char* ws = (char*)d_ws;
  const size_t MB = 1024 * 1024;
  bf16* xb     = (bf16*)(ws + 0 * MB);    // 16 MB: 8192x1024 bf16
  bf16* wqkvb  = (bf16*)(ws + 16 * MB);   //  6 MB: 3072x1024 bf16
  bf16* wprojb = (bf16*)(ws + 22 * MB);   //  2 MB: 1024x1024 bf16
  bf16* Qb     = (bf16*)(ws + 24 * MB);   // 16 MB: [B,H,N,64] bf16 (pre-scaled)
  bf16* Kb     = (bf16*)(ws + 40 * MB);   // 16 MB
  bf16* Vb     = (bf16*)(ws + 56 * MB);   // 16 MB
  bf16* Ob     = (bf16*)(ws + 72 * MB);   // 16 MB: [B,N,1024] bf16

  {
    int n8 = ROWS * DIM_ / 8;
    cvt_f32_bf16<<<(n8 + 255) / 256, 256, 0, stream>>>(x, xb, n8);
  }
  {
    int n8 = NQKV * DIM_ / 8;
    cvt_f32_bf16<<<(n8 + 255) / 256, 256, 0, stream>>>(w_qkv, wqkvb, n8);
  }
  {
    int n8 = DIM_ * DIM_ / 8;
    cvt_f32_bf16<<<(n8 + 255) / 256, 256, 0, stream>>>(w_proj, wprojb, n8);
  }

  qkv_gemm<<<dim3(ROWS / 64, NQKV / 256), 256, 0, stream>>>(xb, wqkvb, b_qkv,
                                                            Qb, Kb, Vb);
  attn_kernel<<<dim3(NSEQ / 64, BATCH * NHEAD), 128, 0, stream>>>(Qb, Kb, Vb, Ob);
  proj_gemm<<<dim3(ROWS / 64, DIM_ / 256), 256, 0, stream>>>(Ob, wprojb, b_proj,
                                                             out);
}
